// SeparableSphericalConvolution_67577015435585
// MI455X (gfx1250) — matrix operand order repaired and checked by reading, not compiled, not run on hardware
//
#include <hip/hip_runtime.h>
#include <hip/hip_bf16.h>

// ---------------------------------------------------------------------------
// Equivariant edge-MLP tensor-product layer for MI455X (gfx1250, wave32).
// Heavy GEMMs (edge MLP 16->64->576) run as f16 v_wmma_f32_16x16x32_f16.
// Tensor product + scatter stay f32 (VALU + global_atomic_add_f32).
// ---------------------------------------------------------------------------

typedef _Float16 v16h  __attribute__((ext_vector_type(16)));
typedef _Float16 half8 __attribute__((ext_vector_type(8)));
typedef float    v8f   __attribute__((ext_vector_type(8)));

#define MUL0   16
#define MUL1   8
#define DIM    40      // MUL0 + 3*MUL1
#define EFD    16
#define HID    64
#define WNUM   576
#define TILE_E 128
#define NTHR   256

static constexpr float INV_SQRT3 = 0.57735026918962576451f;
static constexpr float ALPHA_C   = 0.20412414523193150818f;  // 1/sqrt(24)
static constexpr float INV_SQ8   = 0.35355339059327376220f;  // 1/sqrt(8)
static constexpr float EPS_C     = 1e-5f;

static __device__ __forceinline__ v8f wmma16(v16h a, v16h b, v8f c) {
  // D = A(16x32 f16) * B(32x16 f16) + C(16x16 f32)
  return __builtin_amdgcn_wmma_f32_16x16x32_f16(
      /*neg_a=*/false, a, /*neg_b=*/false, b,
      /*c_mod=*/(short)0, c, /*reuse_a=*/false, /*reuse_b=*/false);
}

// A fragment: lane holds row m, halfs [p .. p+7] and [p+16 .. p+23] (ISA 7.12.2)
static __device__ __forceinline__ v16h packA(const _Float16* p) {
  half8 lo = *(const half8*)(p);
  half8 hi = *(const half8*)(p + 16);
  v16h r;
#pragma unroll
  for (int i = 0; i < 8; ++i) { r[i] = lo[i]; r[i + 8] = hi[i]; }
  return r;
}

// fast reciprocal: single v_rcp_f32 (1 ULP), avoids IEEE div-scale chain
static __device__ __forceinline__ float frcp(float x) {
  return __builtin_amdgcn_rcpf(x);
}

static __device__ __forceinline__ float silu(float x) {
  return x * frcp(1.0f + __expf(-x));
}

// ---------------------------------------------------------------------------
// Kernel 1: per-edge MLP (WMMA) fused with tensor product + scatter-add.
// One workgroup = 128 edges, 256 threads = 8 waves.
// ---------------------------------------------------------------------------
__global__ __launch_bounds__(NTHR) void edge_kernel(
    const float* __restrict__ h, const int* __restrict__ eidx,
    const float* __restrict__ esh, const float* __restrict__ ef,
    const float* __restrict__ W1, const float* __restrict__ b1,
    const float* __restrict__ W2, const float* __restrict__ b2,
    float* __restrict__ agg, float* __restrict__ cnt, int E)
{
  __shared__ alignas(32) _Float16 sW1T[HID * 32];     // B^T for GEMM1, K padded to 32
  __shared__ alignas(32) _Float16 sXA [TILE_E * 32];  // A for GEMM1 (f16, K padded)
  __shared__ alignas(32) _Float16 sH  [TILE_E * HID]; // silu hidden (f16), A for GEMM2
  __shared__ alignas(32) _Float16 sW2T[64 * 64];      // B^T chunk for GEMM2
  __shared__ float sEW [TILE_E * 64];                 // EW chunk (f32)
  __shared__ float sHS [TILE_E * DIM];                // gathered h[src]
  __shared__ float sSH [TILE_E * 4];                  // edge_sh
  __shared__ float sDOT[MUL1 * TILE_E];               // <v_in, sh1>/sqrt3
  __shared__ float sMSG[DIM * TILE_E];                // message accumulator
  __shared__ float sB1[HID];
  __shared__ float sB2[WNUM];
  __shared__ int   sDST[TILE_E];

  const int t    = threadIdx.x;
  const int lane = t & 31;
  const int wave = t >> 5;
  const int e0   = blockIdx.x * TILE_E;

  // ---- stage weights / biases -------------------------------------------
  for (int i = t; i < HID * 32; i += NTHR) {
    int n = i >> 5, k = i & 31;
    sW1T[n * 32 + k] = (k < EFD) ? (_Float16)W1[k * HID + n] : (_Float16)0.f;
  }
  for (int i = t; i < HID;  i += NTHR) sB1[i] = b1[i];
  for (int i = t; i < WNUM; i += NTHR) sB2[i] = b2[i];

  // ---- stage edge features (A matrix, K padded 16->32) ------------------
  for (int i = t; i < TILE_E * 16; i += NTHR) {
    int e = i >> 4, k = i & 15;
    bool valid = (e0 + e) < E;
    sXA[e * 32 + k]      = valid ? (_Float16)ef[(e0 + e) * EFD + k] : (_Float16)0.f;
    sXA[e * 32 + 16 + k] = (_Float16)0.f;
  }

  // ---- stage edge metadata + gather h[src] ------------------------------
  for (int i = t; i < TILE_E; i += NTHR)
    sDST[i] = ((e0 + i) < E) ? eidx[E + e0 + i] : -1;
  for (int i = t; i < TILE_E * 4; i += NTHR) {
    int e = i >> 2, m = i & 3;
    sSH[i] = ((e0 + e) < E) ? esh[(e0 + e) * 4 + m] : 0.f;
  }
  for (int i = t; i < TILE_E * DIM; i += NTHR) {
    int e = i / DIM, j = i - e * DIM;
    bool valid = (e0 + e) < E;
    int s = valid ? eidx[e0 + e] : 0;
    sHS[e * DIM + j] = valid ? h[(long)s * DIM + j] : 0.f;
  }
  for (int i = t; i < DIM * TILE_E; i += NTHR) sMSG[i] = 0.f;
  __syncthreads();

  // ---- dot[u] = <v_in[u,:], sh1> / sqrt(3) ------------------------------
  for (int i = t; i < TILE_E * MUL1; i += NTHR) {
    int e = i >> 3, u = i & 7;
    const float* hs = &sHS[e * DIM + MUL0 + u * 3];
    float d = hs[0] * sSH[e * 4 + 1] + hs[1] * sSH[e * 4 + 2] + hs[2] * sSH[e * 4 + 3];
    sDOT[u * TILE_E + e] = d * INV_SQRT3;
  }

  // ---- GEMM1: (128x32 f16) x (32x64 f16) -> silu -> sH (f16) ------------
#pragma unroll
  for (int tile = 0; tile < 4; ++tile) {            // 32 tiles / 8 waves
    int tl = wave + tile * 8;
    int mt = tl >> 2, nt = tl & 3;
    int m  = lane & 15;
    int kb = (lane < 16) ? 0 : 8;
    v16h A = packA(&sXA[(mt * 16 + m) * 32 + kb]);
    v16h B = packA(&sW1T[(nt * 16 + (lane & 15)) * 32 + ((lane < 16) ? 0 : 8)]);
    v8f  C = {};
    C = wmma16(A, B, C);
    int n  = nt * 16 + (lane & 15);
    int m0 = mt * 16 + ((lane >= 16) ? 8 : 0);
    float bb = sB1[n];
#pragma unroll
    for (int r = 0; r < 8; ++r)
      sH[(m0 + r) * HID + n] = (_Float16)silu(C[r] + bb);
  }
  __syncthreads();

  // ---- GEMM2 in 9 column chunks of 64, fused tensor product -------------
  for (int c = 0; c < 9; ++c) {
    // stage W2^T chunk (cols [64c, 64c+64), K=64) — L2-resident
    for (int i = t; i < 64 * 64; i += NTHR) {
      int k = i >> 6, n = i & 63;
      sW2T[n * 64 + k] = (_Float16)W2[k * WNUM + c * 64 + n];
    }
    if (c + 1 < 9 && t < 64)
      __builtin_prefetch(&W2[t * WNUM + (c + 1) * 64], 0, 1);  // global_prefetch_b8
    __syncthreads();

#pragma unroll
    for (int tile = 0; tile < 4; ++tile) {          // 32 tiles / 8 waves
      int tl = wave + tile * 8;
      int mt = tl >> 2, nt = tl & 3;
      int m  = lane & 15;
      v8f C = {};
#pragma unroll
      for (int kk = 0; kk < 2; ++kk) {              // K = 64 = 2 x 32
        int kb = kk * 32 + ((lane < 16) ? 0 : 8);
        v16h A = packA(&sH[(mt * 16 + m) * HID + kb]);
        v16h B = packA(&sW2T[(nt * 16 + (lane & 15)) * 64 +
                             kk * 32 + ((lane < 16) ? 0 : 8)]);
        C = wmma16(A, B, C);
      }
      int n  = nt * 16 + (lane & 15);
      int m0 = mt * 16 + ((lane >= 16) ? 8 : 0);
      float bb = sB2[c * 64 + n];
#pragma unroll
      for (int r = 0; r < 8; ++r)
        sEW[(m0 + r) * 64 + n] = C[r] + bb;
    }
    __syncthreads();

    // fused tensor-product partials for this chunk.
    // chunk->weight-block map: c=0..3: w1 rows u=4c..; c=4..5: w2 rows u=8(c-4)..;
    // c=6: w3 (all); c=7..8: w4 rows u=4(c-7)..
    for (int i = t; i < DIM * TILE_E; i += NTHR) {
      int j = i >> 7;          // msg component 0..39
      int e = i & (TILE_E - 1);
      const float* ew = &sEW[e * 64];
      float acc = 0.f;
      if (c < 4) {                       // w1: out0[j] += w1[u,j]*s_in[u]*sh0
        if (j < MUL0) {
          float sh0 = sSH[e * 4];
#pragma unroll
          for (int du = 0; du < 4; ++du)
            acc += ew[du * 16 + j] * sHS[e * DIM + 4 * c + du];
          acc *= sh0;
        }
      } else if (c < 6) {                // w2: out1[w,m] += w2[u,w]*s_in[u]*sh1[m]
        if (j >= MUL0) {
          int w = (j - MUL0) / 3, m = (j - MUL0) % 3;
          int u0 = 8 * (c - 4);
          float s = 0.f;
#pragma unroll
          for (int du = 0; du < 8; ++du)
            s += ew[du * 8 + w] * sHS[e * DIM + u0 + du];
          acc = s * sSH[e * 4 + 1 + m];
        }
      } else if (c == 6) {               // w3: out1[w,m] += w3[u,w]*v_in[u,m]*sh0
        if (j >= MUL0) {
          int w = (j - MUL0) / 3, m = (j - MUL0) % 3;
#pragma unroll
          for (int du = 0; du < 8; ++du)
            acc += ew[du * 8 + w] * sHS[e * DIM + MUL0 + du * 3 + m];
          acc *= sSH[e * 4];
        }
      } else {                           // w4: out0[j] += w4[u,j]*dot[u]
        if (j < MUL0) {
          int u0 = 4 * (c - 7);
#pragma unroll
          for (int du = 0; du < 4; ++du)
            acc += ew[du * 16 + j] * sDOT[(u0 + du) * TILE_E + e];
        }
      }
      sMSG[j * TILE_E + e] += acc;
    }
    __syncthreads();
  }

  // ---- scatter: agg[dst] += ALPHA*msg ; cnt[dst] += 1 -------------------
  for (int i = t; i < DIM * TILE_E; i += NTHR) {
    int j = i >> 7, e = i & (TILE_E - 1);
    int d = sDST[e];
    if (d >= 0)
      atomicAdd(&agg[(long)d * DIM + j], ALPHA_C * sMSG[j * TILE_E + e]);
  }
  if (t < TILE_E && sDST[t] >= 0)
    atomicAdd(&cnt[sDST[t]], 1.0f);
}

// ---------------------------------------------------------------------------
// Kernel 2: pre-norm = agg/max(cnt,1) + self-interaction; accumulate stats.
// stats[0..15]=sum(s), [16..31]=sum(s^2), [32..39]=sum over (n,m) of v^2
// ---------------------------------------------------------------------------
__global__ __launch_bounds__(NTHR) void node_pre(
    const float* __restrict__ h, const float* __restrict__ agg,
    const float* __restrict__ cnt, const float* __restrict__ Ws,
    const float* __restrict__ Wv, float* __restrict__ pre,
    float* __restrict__ stats, int N)
{
  __shared__ float sWs[MUL0 * MUL0];
  __shared__ float sWv[MUL1 * MUL1];
  __shared__ float sSum[DIM];

  const int t = threadIdx.x;
  if (t < MUL0 * MUL0) sWs[t] = Ws[t];
  if (t < MUL1 * MUL1) sWv[t] = Wv[t];
  if (t < DIM) sSum[t] = 0.f;
  __syncthreads();

  const int n = blockIdx.x * NTHR + t;
  if (n < N) {
    float hv[DIM];
#pragma unroll
    for (int j = 0; j < DIM; ++j) hv[j] = h[(long)n * DIM + j];
    float inv = frcp(fmaxf(cnt[n], 1.0f));

    float pr[DIM];
#pragma unroll
    for (int w = 0; w < MUL0; ++w) {
      float si = 0.f;
#pragma unroll
      for (int u = 0; u < MUL0; ++u) si += hv[u] * sWs[u * MUL0 + w];
      pr[w] = agg[(long)n * DIM + w] * inv + si * 0.25f;   // /sqrt(16)
    }
#pragma unroll
    for (int w = 0; w < MUL1; ++w)
#pragma unroll
      for (int m = 0; m < 3; ++m) {
        float si = 0.f;
#pragma unroll
        for (int u = 0; u < MUL1; ++u) si += hv[MUL0 + u * 3 + m] * sWv[u * MUL1 + w];
        int j = MUL0 + w * 3 + m;
        pr[j] = agg[(long)n * DIM + j] * inv + si * INV_SQ8;
      }

#pragma unroll
    for (int j = 0; j < DIM; ++j) pre[(long)n * DIM + j] = pr[j];

#pragma unroll
    for (int w = 0; w < MUL0; ++w) {
      atomicAdd(&sSum[w], pr[w]);                 // ds_add_f32
      atomicAdd(&sSum[MUL0 + w], pr[w] * pr[w]);
    }
#pragma unroll
    for (int w = 0; w < MUL1; ++w) {
      float q = 0.f;
#pragma unroll
      for (int m = 0; m < 3; ++m) { float v = pr[MUL0 + w * 3 + m]; q += v * v; }
      atomicAdd(&sSum[32 + w], q);
    }
  }
  __syncthreads();
  if (t < DIM) atomicAdd(&stats[t], sSum[t]);
}

// ---------------------------------------------------------------------------
// Kernel 3: finalize normalization + residual.
// ---------------------------------------------------------------------------
__global__ __launch_bounds__(NTHR) void node_final(
    const float* __restrict__ h, const float* __restrict__ pre,
    const float* __restrict__ stats, const float* __restrict__ gs,
    const float* __restrict__ bs, const float* __restrict__ gv,
    float* __restrict__ out, int N)
{
  __shared__ float sMean[MUL0], sScaleS[MUL0], sBeta[MUL0], sScaleV[MUL1];
  const int t = threadIdx.x;
  const float invN = frcp((float)N);
  if (t < MUL0) {
    float mu  = stats[t] * invN;
    float var = stats[MUL0 + t] * invN - mu * mu;
    sMean[t]   = mu;
    sScaleS[t] = gs[t] * frcp(sqrtf(var + EPS_C));
    sBeta[t]   = bs[t];
  }
  if (t < MUL1) {
    float ms = stats[32 + t] * invN * (1.0f / 3.0f);
    sScaleV[t] = gv[t] * frcp(sqrtf(ms + EPS_C));
  }
  __syncthreads();

  const int n = blockIdx.x * NTHR + t;
  if (n < N) {
#pragma unroll
    for (int w = 0; w < MUL0; ++w) {
      float s = pre[(long)n * DIM + w];
      out[(long)n * DIM + w] =
          (s - sMean[w]) * sScaleS[w] + sBeta[w] + h[(long)n * DIM + w];
    }
#pragma unroll
    for (int w = 0; w < MUL1; ++w)
#pragma unroll
      for (int m = 0; m < 3; ++m) {
        int j = MUL0 + w * 3 + m;
        out[(long)n * DIM + j] =
            pre[(long)n * DIM + j] * sScaleV[w] + h[(long)n * DIM + j];
      }
  }
}

// ---------------------------------------------------------------------------
extern "C" void kernel_launch(void* const* d_in, const int* in_sizes, int n_in,
                              void* d_out, int out_size, void* d_ws, size_t ws_size,
                              hipStream_t stream) {
  const float* h   = (const float*)d_in[0];
  const int*   eix = (const int*)  d_in[1];
  const float* esh = (const float*)d_in[2];
  const float* ef  = (const float*)d_in[3];
  const float* W1  = (const float*)d_in[4];
  const float* b1  = (const float*)d_in[5];
  const float* W2  = (const float*)d_in[6];
  const float* b2  = (const float*)d_in[7];
  const float* Ws  = (const float*)d_in[8];
  const float* Wv  = (const float*)d_in[9];
  const float* gs  = (const float*)d_in[10];
  const float* bs  = (const float*)d_in[11];
  const float* gv  = (const float*)d_in[12];
  float* out = (float*)d_out;

  const int N = in_sizes[0] / DIM;   // 50000
  const int E = in_sizes[1] / 2;     // 400000

  // workspace: agg[N*40] | cnt[N] | pre[N*40] | stats[40]
  float* agg   = (float*)d_ws;
  float* cnt   = agg + (size_t)N * DIM;
  float* pre   = cnt + N;
  float* stats = pre + (size_t)N * DIM;

  hipMemsetAsync(agg, 0, sizeof(float) * ((size_t)N * (DIM + 1)), stream); // agg+cnt
  hipMemsetAsync(stats, 0, sizeof(float) * DIM, stream);

  const int eblocks = (E + TILE_E - 1) / TILE_E;   // 3125
  edge_kernel<<<eblocks, NTHR, 0, stream>>>(h, eix, esh, ef, W1, b1, W2, b2,
                                            agg, cnt, E);

  const int nblocks = (N + NTHR - 1) / NTHR;       // 196
  node_pre<<<nblocks, NTHR, 0, stream>>>(h, agg, cnt, Ws, Wv, pre, stats, N);
  node_final<<<nblocks, NTHR, 0, stream>>>(h, pre, stats, gs, bs, gv, out, N);
}
